// RWKV_TimeMix_89807766159667
// MI455X (gfx1250) — matrix operand order repaired and checked by reading, not compiled, not run on hardware
//
#include <hip/hip_runtime.h>
#include <hip/hip_bf16.h>

#define B_   4
#define T_   1024
#define C_   960
#define H_   15
#define N_   64
#define BT_  4096

typedef __attribute__((ext_vector_type(16))) __bf16 v16bf;
typedef __attribute__((ext_vector_type(8)))  float  v8f;

#define BMT 128
#define BNT 128
#define BKT 32
#define LDK (BKT + 8)   // 40 bf16 = 80 B row stride (16B aligned, 20-bank stagger)
#define GT  256

union PK4 { __bf16 b[4]; uint2 u; };

// ---- fragment loaders (ISA 7.12.2 16-bit layouts, wave32) ----
// As[row][k] row-major. Lane<16: K={0..7,16..23}; lane>=16: K={8..15,24..31}
__device__ inline v16bf load_a_frag(const __bf16* As, int row, int lane) {
  const __bf16* p = As + row * LDK + ((lane < 16) ? 0 : 8);
  union { uint4 u[2]; v16bf v; } t;
  t.u[0] = *(const uint4*)(p);
  t.u[1] = *(const uint4*)(p + 16);
  return t.v;
}
// Bs[col][k] (K-contiguous per column). Lane<16: K={0..7,16..23}; lane>=16: K={8..15,24..31}
__device__ inline v16bf load_b_frag(const __bf16* Bs, int col, int lane) {
  const __bf16* p = Bs + col * LDK + ((lane < 16) ? 0 : 8);
  union { uint4 u[2]; v16bf v; } t;
  t.u[0] = *(const uint4*)(p);
  t.u[1] = *(const uint4*)(p + 16);
  return t.v;
}

// ---- fused (token-shift-mix) -> bf16 WMMA GEMM -> epilogue ----
// Preconditions (guaranteed by all launches below): M % BMT == 0, K % BKT == 0.
// EPI: 0=none, 1=tanh, 2=sigmoid. MIXED: A[m,:] = x[m] + (x[m-1]-x[m])*mix, x[-1]=0 per T-segment.
template <int EPI, bool MIXED>
__global__ __launch_bounds__(GT) void wmma_gemm(
    const float* __restrict__ A, const float* __restrict__ mix,
    const float* __restrict__ W, const float* __restrict__ bias,
    float* __restrict__ out, int M, int K, int Nc)
{
  __shared__ __align__(16) __bf16 As[BMT][LDK];
  __shared__ __align__(16) __bf16 Bs[BNT][LDK];

  const int tid  = threadIdx.x;
  const int lane = tid & 31;
  const int wave = tid >> 5;
  const int wm   = wave & 1;   // 2 wave-rows of 64
  const int wn   = wave >> 1;  // 4 wave-cols of 32
  const int bM   = blockIdx.y * BMT;
  const int bN   = blockIdx.x * BNT;

  // B micro-tile coords: one 4x4 tile per thread (8 K-groups x 32 col-groups)
  const int bkr  = (tid >> 5) * 4;
  const int bcol = (tid & 31) * 4;

  v8f acc[4][2];
#pragma unroll
  for (int i = 0; i < 4; ++i)
#pragma unroll
    for (int j = 0; j < 2; ++j)
#pragma unroll
      for (int e = 0; e < 8; ++e) acc[i][j][e] = 0.0f;

  for (int k0 = 0; k0 < K; k0 += BKT) {
    // ---- A tile: 128x32, float4 global loads, packed ds_store_b64, mix fused ----
#pragma unroll
    for (int l = 0; l < 4; ++l) {
      int chunk = l * GT + tid;
      int row = chunk >> 3, kc = (chunk & 7) * 4;
      int gm = bM + row;       // < M always
      int gk = k0 + kc;        // < K always
      float4 cur = *(const float4*)(A + (size_t)gm * K + gk);
      float4 v;
      if (MIXED) {
        float4 prev = make_float4(0.f, 0.f, 0.f, 0.f);
        if ((gm & (T_ - 1)) != 0)
          prev = *(const float4*)(A + (size_t)(gm - 1) * K + gk);
        float4 mx = *(const float4*)(mix + gk);
        v.x = cur.x + (prev.x - cur.x) * mx.x;
        v.y = cur.y + (prev.y - cur.y) * mx.y;
        v.z = cur.z + (prev.z - cur.z) * mx.z;
        v.w = cur.w + (prev.w - cur.w) * mx.w;
      } else {
        v = cur;
      }
      PK4 pk;
      pk.b[0] = (__bf16)v.x; pk.b[1] = (__bf16)v.y;
      pk.b[2] = (__bf16)v.z; pk.b[3] = (__bf16)v.w;
      *(uint2*)(&As[row][kc]) = pk.u;
    }

    // ---- B tile: 32x128, 4x4 register transpose, packed ds_store_b64 ----
    {
      int gk = k0 + bkr;       // < K always
      int gn = bN + bcol;
      float rr[4][4];
#pragma unroll
      for (int j = 0; j < 4; ++j)
#pragma unroll
        for (int c = 0; c < 4; ++c) rr[j][c] = 0.0f;
      if (gn + 3 < Nc) {
#pragma unroll
        for (int j = 0; j < 4; ++j) {
          float4 t4 = *(const float4*)(W + (size_t)(gk + j) * Nc + gn);
          rr[j][0] = t4.x; rr[j][1] = t4.y; rr[j][2] = t4.z; rr[j][3] = t4.w;
        }
      } else if (gn < Nc) {
#pragma unroll
        for (int j = 0; j < 4; ++j)
          for (int c = 0; c < 4; ++c)
            if (gn + c < Nc) rr[j][c] = W[(size_t)(gk + j) * Nc + gn + c];
      }
#pragma unroll
      for (int c = 0; c < 4; ++c) {
        PK4 pk;
        pk.b[0] = (__bf16)rr[0][c]; pk.b[1] = (__bf16)rr[1][c];
        pk.b[2] = (__bf16)rr[2][c]; pk.b[3] = (__bf16)rr[3][c];
        *(uint2*)(&Bs[bcol + c][bkr]) = pk.u;
      }
    }
    __syncthreads();

    v16bf af[4], bf[2];
#pragma unroll
    for (int nt = 0; nt < 2; ++nt)
      bf[nt] = load_b_frag(&Bs[0][0], wn * 32 + nt * 16 + (lane & 15), lane);
#pragma unroll
    for (int mt = 0; mt < 4; ++mt)
      af[mt] = load_a_frag(&As[0][0], wm * 64 + mt * 16 + (lane & 15), lane);
#pragma unroll
    for (int mt = 0; mt < 4; ++mt)
#pragma unroll
      for (int nt = 0; nt < 2; ++nt)
        acc[mt][nt] = __builtin_amdgcn_wmma_f32_16x16x32_bf16(
            false, af[mt], false, bf[nt], (short)0, acc[mt][nt], false, false);
    __syncthreads();
  }

  // C/D layout: reg e, lane l -> M = e + 8*(l>=16), N = l&15
  const int ml = (lane >> 4) * 8;
  const int nl = lane & 15;
#pragma unroll
  for (int mt = 0; mt < 4; ++mt)
#pragma unroll
    for (int nt = 0; nt < 2; ++nt) {
      int gn = bN + wn * 32 + nt * 16 + nl;
      if (gn < Nc) {
        float bv = bias ? bias[gn] : 0.0f;
#pragma unroll
        for (int e = 0; e < 8; ++e) {
          int gm = bM + wm * 64 + mt * 16 + ml + e;   // < M always
          float c = acc[mt][nt][e] + bv;
          if (EPI == 1) c = tanhf(c);
          else if (EPI == 2) c = 1.0f / (1.0f + __expf(-c));
          out[(size_t)gm * Nc + gn] = c;
        }
      }
    }
}

// ---- prep: v residual mix, kk normalize, k update, wh = sigmoid(w)*exp(-0.5) ----
__global__ __launch_bounds__(256) void prep_kernel(
    const float* __restrict__ v_first,
    const float* __restrict__ k_k, const float* __restrict__ k_a,
    float* __restrict__ kbuf, float* __restrict__ vbuf,
    float* __restrict__ wbuf, const float* __restrict__ abuf,
    float* __restrict__ kkbuf, const float* __restrict__ vmbuf)
{
  const int gid  = blockIdx.x * 8 + (threadIdx.x >> 5);
  const int lane = threadIdx.x & 31;
  const int h = gid % H_, row = gid / H_;
  const size_t base = (size_t)row * C_ + h * N_;
  const int i0 = lane, i1 = lane + 32;

  float vp0 = vbuf[base + i0], vp1 = vbuf[base + i1];
  float vm0 = vmbuf[base + i0], vm1 = vmbuf[base + i1];
  vbuf[base + i0] = vp0 + (v_first[base + i0] - vp0) * vm0;
  vbuf[base + i1] = vp1 + (v_first[base + i1] - vp1) * vm1;

  float kp0 = kbuf[base + i0], kp1 = kbuf[base + i1];
  float kk0 = kp0 * k_k[h * N_ + i0], kk1 = kp1 * k_k[h * N_ + i1];
  float ss = kk0 * kk0 + kk1 * kk1;
#pragma unroll
  for (int m = 16; m >= 1; m >>= 1) ss += __shfl_xor(ss, m, 32);
  float inv = 1.0f / fmaxf(sqrtf(ss), 1e-12f);
  kkbuf[base + i0] = kk0 * inv;
  kkbuf[base + i1] = kk1 * inv;

  float a0v = abuf[base + i0], a1v = abuf[base + i1];
  kbuf[base + i0] = kp0 * (1.0f + (a0v - 1.0f) * k_a[h * N_ + i0]);
  kbuf[base + i1] = kp1 * (1.0f + (a1v - 1.0f) * k_a[h * N_ + i1]);

  float w0v = wbuf[base + i0], w1v = wbuf[base + i1];
  wbuf[base + i0] = 0.60653065971f / (1.0f + __expf(-w0v));
  wbuf[base + i1] = 0.60653065971f / (1.0f + __expf(-w1v));
}

// ---- sequential scan: one block per (b,h), thread i owns state row i in 64 VGPRs ----
__global__ __launch_bounds__(64) void scan_kernel(
    const float* __restrict__ rbuf, const float* __restrict__ wbuf,
    const float* __restrict__ kbuf, const float* __restrict__ vbuf,
    const float* __restrict__ abuf, const float* __restrict__ kkbuf,
    float* __restrict__ wkv)
{
  __shared__ float rL[N_], wL[N_], kL[N_], aL[N_], bL[N_];
  const int i = threadIdx.x;
  const int b = blockIdx.x / H_, h = blockIdx.x % H_;

  float st[N_];
#pragma unroll
  for (int j = 0; j < N_; ++j) st[j] = 0.0f;

  for (int t = 0; t < T_; ++t) {
    const size_t base = ((size_t)(b * T_ + t)) * C_ + h * N_;
    float kkv = kkbuf[base + i];
    rL[i] = rbuf[base + i];
    wL[i] = wbuf[base + i];
    kL[i] = kbuf[base + i];
    aL[i] = -kkv;
    bL[i] = kkv * abuf[base + i];
    float vi = vbuf[base + i];
    __syncthreads();

    float sa = 0.0f;
#pragma unroll
    for (int j = 0; j < N_; ++j) sa += st[j] * aL[j];
    float o = 0.0f;
#pragma unroll
    for (int j = 0; j < N_; ++j) {
      st[j] = st[j] * wL[j] + sa * bL[j] + vi * kL[j];
      o += st[j] * rL[j];
    }
    wkv[base + i] = o;
    __syncthreads();
  }
}

// ---- post: per-head groupnorm + r*k*r_k bonus + gate; gated written over r buffer ----
__global__ __launch_bounds__(256) void post_kernel(
    float* __restrict__ rbuf, const float* __restrict__ kbuf,
    const float* __restrict__ vbuf, const float* __restrict__ gbuf,
    const float* __restrict__ wkv,
    const float* __restrict__ r_k, const float* __restrict__ ln_g,
    const float* __restrict__ ln_b)
{
  const int gid  = blockIdx.x * 8 + (threadIdx.x >> 5);
  const int lane = threadIdx.x & 31;
  const int h = gid % H_, row = gid / H_;
  const size_t base = (size_t)row * C_ + h * N_;
  const int i0 = lane, i1 = lane + 32;

  float w0v = wkv[base + i0], w1v = wkv[base + i1];
  float r0 = rbuf[base + i0], r1 = rbuf[base + i1];
  float k0v = kbuf[base + i0], k1v = kbuf[base + i1];

  float s  = w0v + w1v;
  float ss = w0v * w0v + w1v * w1v;
  float bd = r0 * k0v * r_k[h * N_ + i0] + r1 * k1v * r_k[h * N_ + i1];
#pragma unroll
  for (int m = 16; m >= 1; m >>= 1) {
    s  += __shfl_xor(s, m, 32);
    ss += __shfl_xor(ss, m, 32);
    bd += __shfl_xor(bd, m, 32);
  }
  float mean = s * (1.0f / N_);
  float var  = ss * (1.0f / N_) - mean * mean;
  float rstd = rsqrtf(var + 0.00064f);

  float xn0 = (w0v - mean) * rstd * ln_g[h * N_ + i0] + ln_b[h * N_ + i0];
  float xn1 = (w1v - mean) * rstd * ln_g[h * N_ + i1] + ln_b[h * N_ + i1];
  rbuf[base + i0] = (xn0 + bd * vbuf[base + i0]) * gbuf[base + i0];
  rbuf[base + i1] = (xn1 + bd * vbuf[base + i1]) * gbuf[base + i1];
}

extern "C" void kernel_launch(void* const* d_in, const int* in_sizes, int n_in,
                              void* d_out, int out_size, void* d_ws, size_t ws_size,
                              hipStream_t stream) {
  const float* x       = (const float*)d_in[0];
  const float* v_first = (const float*)d_in[1];
  const float* x_r = (const float*)d_in[2];
  const float* x_w = (const float*)d_in[3];
  const float* x_k = (const float*)d_in[4];
  const float* x_v = (const float*)d_in[5];
  const float* x_a = (const float*)d_in[6];
  const float* x_g = (const float*)d_in[7];
  const float* w0  = (const float*)d_in[8];
  const float* w1  = (const float*)d_in[9];
  const float* w2  = (const float*)d_in[10];
  const float* a0  = (const float*)d_in[11];
  const float* a1  = (const float*)d_in[12];
  const float* a2  = (const float*)d_in[13];
  const float* v0  = (const float*)d_in[14];
  const float* v1  = (const float*)d_in[15];
  const float* v2  = (const float*)d_in[16];
  const float* g1  = (const float*)d_in[17];
  const float* g2  = (const float*)d_in[18];
  const float* k_k = (const float*)d_in[19];
  const float* k_a = (const float*)d_in[20];
  const float* r_k = (const float*)d_in[21];
  const float* Wr  = (const float*)d_in[22];
  const float* Wk  = (const float*)d_in[23];
  const float* Wv  = (const float*)d_in[24];
  const float* Wo  = (const float*)d_in[25];
  const float* ln_g = (const float*)d_in[26];
  const float* ln_b = (const float*)d_in[27];
  float* out = (float*)d_out;

  const size_t nBTC = (size_t)BT_ * C_;
  float* ws   = (float*)d_ws;
  float* rb   = ws;              // r -> later gated(xo*g)
  float* kb   = ws + 1 * nBTC;   // k
  float* vb   = ws + 2 * nBTC;   // v
  float* wb   = ws + 3 * nBTC;   // w_raw -> wh
  float* ab   = ws + 4 * nBTC;   // a
  float* kkb  = ws + 5 * nBTC;   // kk normalized
  float* gb   = ws + 6 * nBTC;   // g
  float* vmb  = ws + 7 * nBTC;   // vmix -> later wkv
  float* tb   = ws + 8 * nBTC;   // low-rank temp (<=128 cols)

  dim3 blk(GT);
  auto grd = [](int M, int Nc) { return dim3((Nc + BNT - 1) / BNT, (M + BMT - 1) / BMT); };
  const int M = BT_;

  // big projections (token-shift mix fused into A load)
  wmma_gemm<0, true><<<grd(M, C_), blk, 0, stream>>>(x, x_r, Wr, nullptr, rb, M, C_, C_);
  wmma_gemm<0, true><<<grd(M, C_), blk, 0, stream>>>(x, x_k, Wk, nullptr, kb, M, C_, C_);
  wmma_gemm<0, true><<<grd(M, C_), blk, 0, stream>>>(x, x_v, Wv, nullptr, vb, M, C_, C_);
  // w = w0 + tanh(xw@w1)@w2
  wmma_gemm<1, true><<<grd(M, 64), blk, 0, stream>>>(x, x_w, w1, nullptr, tb, M, C_, 64);
  wmma_gemm<0, false><<<grd(M, C_), blk, 0, stream>>>(tb, nullptr, w2, w0, wb, M, 64, C_);
  // a = sigmoid(a0 + (xa@a1)@a2)
  wmma_gemm<0, true><<<grd(M, 64), blk, 0, stream>>>(x, x_a, a1, nullptr, tb, M, C_, 64);
  wmma_gemm<2, false><<<grd(M, C_), blk, 0, stream>>>(tb, nullptr, a2, a0, ab, M, 64, C_);
  // vmix = sigmoid(v0 + (xv@v1)@v2)
  wmma_gemm<0, true><<<grd(M, 32), blk, 0, stream>>>(x, x_v, v1, nullptr, tb, M, C_, 32);
  wmma_gemm<2, false><<<grd(M, C_), blk, 0, stream>>>(tb, nullptr, v2, v0, vmb, M, 32, C_);
  // g = sigmoid(xg@g1)@g2
  wmma_gemm<2, true><<<grd(M, 128), blk, 0, stream>>>(x, x_g, g1, nullptr, tb, M, C_, 128);
  wmma_gemm<0, false><<<grd(M, C_), blk, 0, stream>>>(tb, nullptr, g2, nullptr, gb, M, 128, C_);

  // elementwise prep (v residual, kk norm, k update, wh)
  prep_kernel<<<dim3(BT_ * H_ / 8), dim3(256), 0, stream>>>(
      v_first, k_k, k_a, kb, vb, wb, ab, kkb, vmb);

  // sequential state scan; wkv overwrites vmix buffer
  scan_kernel<<<dim3(B_ * H_), dim3(64), 0, stream>>>(rb, wb, kb, vb, ab, kkb, vmb);

  // groupnorm + bonus + gate; gated overwrites r buffer
  post_kernel<<<dim3(BT_ * H_ / 8), dim3(256), 0, stream>>>(
      rb, kb, vb, gb, vmb, r_k, ln_g, ln_b);

  // final projection
  wmma_gemm<0, false><<<grd(M, C_), blk, 0, stream>>>(rb, nullptr, Wo, nullptr, out, M, C_, C_);
}
